// MinLSTM_13984413516110
// MI455X (gfx1250) — matrix operand order repaired and checked by reading, not compiled, not run on hardware
//
#include <hip/hip_runtime.h>

typedef __attribute__((ext_vector_type(16))) __bf16 v16bf;
typedef __attribute__((ext_vector_type(8)))  float  v8f;
typedef unsigned short ushort_t;

#define B_ 16
#define T_ 2048
#define H_ 768
#define L_ 2
#define M_ (B_ * T_)       // 32768 rows in the gate GEMMs
#define EPS_ 1e-8f

// ---------- fp32 -> bf16 (round to nearest even) ----------
__device__ __forceinline__ ushort_t f2bf(float f) {
  unsigned int u = __builtin_bit_cast(unsigned int, f);
  u += 0x7fffu + ((u >> 16) & 1u);
  return (ushort_t)(u >> 16);
}

__global__ __launch_bounds__(256) void conv_f32_bf16(const float* __restrict__ src,
                                                     ushort_t* __restrict__ dst, int n) {
  int i = blockIdx.x * blockDim.x + threadIdx.x;
  if (i < n) dst[i] = f2bf(src[i]);
}

// Pack Wf/Wi/Wh (fp32, [L,H,H]) -> bf16 [L][3][H][H]
__global__ __launch_bounds__(256) void pack_weights(const float* __restrict__ Wf,
                                                    const float* __restrict__ Wi,
                                                    const float* __restrict__ Wh,
                                                    ushort_t* __restrict__ Wb) {
  int i = blockIdx.x * blockDim.x + threadIdx.x;
  const int total = L_ * 3 * H_ * H_;
  if (i >= total) return;
  int k = i % H_;
  int rest = i / H_;
  int n = rest % H_;
  rest /= H_;
  int g = rest % 3;
  int l = rest / 3;
  const float* src = (g == 0) ? Wf : ((g == 1) ? Wi : Wh);
  Wb[i] = f2bf(src[((size_t)l * H_ + n) * H_ + k]);
}

// ---------- WMMA fragment loads (ISA §7.12.2 layouts) ----------
struct BFrag { float4 lo, hi; };   // 32 bytes == v16bf

// A (16-bit, 16x32): lane<16 -> row base+lane, K = {k0..k0+7, k0+16..k0+23}
//                    lane>=16 -> row base+lane-16, K = {k0+8..15, k0+24..31}
__device__ __forceinline__ v16bf lda(const ushort_t* __restrict__ p, int k0) {
  BFrag r;
  r.lo = *reinterpret_cast<const float4*>(p + k0);
  r.hi = *reinterpret_cast<const float4*>(p + k0 + 16);
  return __builtin_bit_cast(v16bf, r);
}

// B (16-bit, 32x16): lane n (0..15) holds W[n][k0..k0+7, k0+16..k0+23]; lanes 16..31 K={k0+8..15, k0+24..31}.
__device__ __forceinline__ v16bf ldb(const ushort_t* __restrict__ p, int k0) {
  BFrag r;
  r.lo = *reinterpret_cast<const float4*>(p + k0);
  r.hi = *reinterpret_cast<const float4*>(p + k0 + 16);
  return __builtin_bit_cast(v16bf, r);
}

#define WMMA_BF16(A, Bm, C) \
  __builtin_amdgcn_wmma_f32_16x16x32_bf16(false, (A), false, (Bm), (short)0, (C), false, false)

// ---------- fused 3-gate GEMM + pointwise epilogue ----------
// grid = (H/32, M/128), block = 256 (8 waves, 4 M-waves x 2 N-waves).
// Each wave: two 16-row M-subtiles x one 16-col N-tile x 3 gates = 6 WMMA / K-slab.
// Manual 2x ping-pong software pipeline: both fragment buffers are written
// directly by loads, so the register rotation needs no v_mov copies.
__global__ __launch_bounds__(256, 2) void gemm_gates(
    const ushort_t* __restrict__ Xb, const ushort_t* __restrict__ Wb,
    const float* __restrict__ bfp, const float* __restrict__ bip,
    const float* __restrict__ bhp,
    float* __restrict__ a_out, float* __restrict__ b_out) {
  const int lane  = threadIdx.x & 31;
  const int wave  = threadIdx.x >> 5;
  const int waveM = wave >> 1;                    // 0..3
  const int waveN = wave & 1;                     // 0..1
  const int m0    = blockIdx.y * 128 + waveM * 32;   // subtiles at m0 and m0+16
  const int nn    = blockIdx.x * 32 + waveN * 16 + (lane & 15);

  const int asel = (lane >= 16) ? 8 : 0;
  const ushort_t* a0p = Xb + (size_t)(m0 + (lane & 15)) * H_ + asel;
  const ushort_t* a1p = a0p + (size_t)16 * H_;
  const int bsel = (lane >= 16) ? 8 : 0;
  const ushort_t* wfp = Wb + (size_t)nn * H_ + bsel;
  const ushort_t* wip = wfp + (size_t)H_ * H_;
  const ushort_t* whp = wip + (size_t)H_ * H_;

  v8f cf0 = {}, ci0 = {}, ch0 = {};
  v8f cf1 = {}, ci1 = {}, ch1 = {};

  // ---- pipeline prologue: buffer A <- slab 0 ----
  v16bf a0A = lda(a0p, 0);
  v16bf a1A = lda(a1p, 0);
  v16bf wfA = ldb(wfp, 0);
  v16bf wiA = ldb(wip, 0);
  v16bf whA = ldb(whp, 0);

  v16bf a0B, a1B, wfB, wiB, whB;

  // 24 K-slabs total: loop covers slabs 0..21 (11 iterations x 2 slabs)
#pragma unroll 1
  for (int k0 = 0; k0 < (H_ - 64); k0 += 64) {
    // buffer B <- slab (k0/32 + 1)
    a0B = lda(a0p, k0 + 32);
    a1B = lda(a1p, k0 + 32);
    wfB = ldb(wfp, k0 + 32);
    wiB = ldb(wip, k0 + 32);
    whB = ldb(whp, k0 + 32);

    // consume buffer A
    cf0 = WMMA_BF16(a0A, wfA, cf0);
    cf1 = WMMA_BF16(a1A, wfA, cf1);
    ci0 = WMMA_BF16(a0A, wiA, ci0);
    ci1 = WMMA_BF16(a1A, wiA, ci1);
    ch0 = WMMA_BF16(a0A, whA, ch0);
    ch1 = WMMA_BF16(a1A, whA, ch1);

    // buffer A <- slab (k0/32 + 2)
    a0A = lda(a0p, k0 + 64);
    a1A = lda(a1p, k0 + 64);
    wfA = ldb(wfp, k0 + 64);
    wiA = ldb(wip, k0 + 64);
    whA = ldb(whp, k0 + 64);

    // consume buffer B
    cf0 = WMMA_BF16(a0B, wfB, cf0);
    cf1 = WMMA_BF16(a1B, wfB, cf1);
    ci0 = WMMA_BF16(a0B, wiB, ci0);
    ci1 = WMMA_BF16(a1B, wiB, ci1);
    ch0 = WMMA_BF16(a0B, whB, ch0);
    ch1 = WMMA_BF16(a1B, whB, ch1);
  }

  // ---- tail: buffer A holds slab 22; load slab 23 into B, drain both ----
  a0B = lda(a0p, H_ - 32);
  a1B = lda(a1p, H_ - 32);
  wfB = ldb(wfp, H_ - 32);
  wiB = ldb(wip, H_ - 32);
  whB = ldb(whp, H_ - 32);

  cf0 = WMMA_BF16(a0A, wfA, cf0);
  cf1 = WMMA_BF16(a1A, wfA, cf1);
  ci0 = WMMA_BF16(a0A, wiA, ci0);
  ci1 = WMMA_BF16(a1A, wiA, ci1);
  ch0 = WMMA_BF16(a0A, whA, ch0);
  ch1 = WMMA_BF16(a1A, whA, ch1);

  cf0 = WMMA_BF16(a0B, wfB, cf0);
  cf1 = WMMA_BF16(a1B, wfB, cf1);
  ci0 = WMMA_BF16(a0B, wiB, ci0);
  ci1 = WMMA_BF16(a1B, wiB, ci1);
  ch0 = WMMA_BF16(a0B, whB, ch0);
  ch1 = WMMA_BF16(a1B, whB, ch1);

  // ---- pointwise epilogue ----
  // C/D layout: VGPR r, lane -> row = msub + r + (lane>=16 ? 8 : 0), col = nn
  const float bfv = bfp[nn];
  const float biv = bip[nn];
  const float bhv = bhp[nn];
  const int r0 = m0 + ((lane >= 16) ? 8 : 0);
#pragma unroll
  for (int r = 0; r < 8; ++r) {
    {
      float f  = 1.0f / (1.0f + __expf(-(cf0[r] + bfv)));
      float i  = 1.0f / (1.0f + __expf(-(ci0[r] + biv)));
      float ht = ch0[r] + bhv;
      float inv = 1.0f / (f + i + EPS_);
      size_t idx = (size_t)(r0 + r) * H_ + nn;
      a_out[idx] = f * inv;
      b_out[idx] = (i * inv) * ht;
    }
    {
      float f  = 1.0f / (1.0f + __expf(-(cf1[r] + bfv)));
      float i  = 1.0f / (1.0f + __expf(-(ci1[r] + biv)));
      float ht = ch1[r] + bhv;
      float inv = 1.0f / (f + i + EPS_);
      size_t idx = (size_t)(r0 + 16 + r) * H_ + nn;
      a_out[idx] = f * inv;
      b_out[idx] = (i * inv) * ht;
    }
  }
}

// ---------- linear recurrence h_t = a_t * h_{t-1} + b_t ----------
__global__ __launch_bounds__(256) void scan_kernel(
    const float* __restrict__ av, const float* __restrict__ bv,
    const float* __restrict__ h0, int layer,
    float* __restrict__ hout, float* __restrict__ hlast) {
  int tid = blockIdx.x * blockDim.x + threadIdx.x;  // 0 .. B*H-1
  int b = tid / H_;
  int h = tid - b * H_;
  float hv = h0[((size_t)b * L_ + layer) * H_ + h];
  size_t idx = (size_t)b * T_ * H_ + h;
#pragma unroll 4
  for (int t = 0; t < T_; ++t, idx += H_) {
    hv = fmaf(av[idx], hv, bv[idx]);
    hout[idx] = hv;
  }
  hlast[((size_t)b * L_ + layer) * H_ + h] = hv;
}

// ---------- host side ----------
extern "C" void kernel_launch(void* const* d_in, const int* in_sizes, int n_in,
                              void* d_out, int out_size, void* d_ws, size_t ws_size,
                              hipStream_t stream) {
  (void)in_sizes; (void)n_in; (void)out_size; (void)ws_size;
  const float* x  = (const float*)d_in[0];
  const float* h0 = (const float*)d_in[1];
  const float* Wf = (const float*)d_in[2];
  const float* bf = (const float*)d_in[3];
  const float* Wi = (const float*)d_in[4];
  const float* bi = (const float*)d_in[5];
  const float* Wh = (const float*)d_in[6];
  const float* bh = (const float*)d_in[7];

  float* out   = (float*)d_out;                 // [B,T,H] layer-2 hidden states
  float* hlast = out + (size_t)M_ * H_;         // [B,L,H] last h per layer

  // Workspace partition (~259 MB total)
  char* p = (char*)d_ws;
  ushort_t* Xb = (ushort_t*)p;  p += (size_t)M_ * H_ * sizeof(ushort_t);          // 48 MB
  ushort_t* Wb = (ushort_t*)p;  p += (size_t)L_ * 3 * H_ * H_ * sizeof(ushort_t); // 6.75 MB
  float* a_buf = (float*)p;     p += (size_t)M_ * H_ * sizeof(float);             // 96 MB
  float* b_buf = (float*)p;                                                       // 96 MB

  const int nx = M_ * H_;              // 25,165,824
  const int nw = L_ * 3 * H_ * H_;     // 3,538,944

  pack_weights<<<(nw + 255) / 256, 256, 0, stream>>>(Wf, Wi, Wh, Wb);
  conv_f32_bf16<<<(nx + 255) / 256, 256, 0, stream>>>(x, Xb, nx);

  dim3 gg(H_ / 32, M_ / 128);   // (24, 256)

  // ---- layer 0 ----
  gemm_gates<<<gg, 256, 0, stream>>>(Xb, Wb, bf, bi, bh, a_buf, b_buf);
  scan_kernel<<<(B_ * H_) / 256, 256, 0, stream>>>(a_buf, b_buf, h0, 0, b_buf, hlast);
  conv_f32_bf16<<<(nx + 255) / 256, 256, 0, stream>>>(b_buf, Xb, nx);

  // ---- layer 1 ----
  gemm_gates<<<gg, 256, 0, stream>>>(Xb, Wb + (size_t)3 * H_ * H_,
                                     bf + H_, bi + H_, bh + H_, a_buf, b_buf);
  scan_kernel<<<(B_ * H_) / 256, 256, 0, stream>>>(a_buf, b_buf, h0, 1, out, hlast);
}
